// WindowGNN_74603581931881
// MI455X (gfx1250) — compile-verified, matrix operand order checked by reading, not run on hardware
//
#include <hip/hip_runtime.h>
#include <hip/hip_bf16.h>

// ---------------------------------------------------------------------------
// Problem constants (from reference)
// ---------------------------------------------------------------------------
#define NN   10000
#define EE_  160000
#define DD   512
#define OUTD 250
#define OUTP 256            // OUTD padded to multiple of 16 (B rows zero-filled)
#define LL   4
#define EEP  (EE_ + NN)     // edges + self loops
#define SLOPE 0.2f

// GEMM tiling
#define BK        128                    // K-chunk staged in LDS
#define COLS_BLK  128                    // columns per block
#define COL_LDS   136                    // 128 data shorts + 8 pad shorts (16B per 256B -> bank skew 4)
#define STAGE_SH  (COLS_BLK * COL_LDS)   // shorts per LDS stage buffer
#define LDS_BYTES (2 * STAGE_SH * 2)     // double buffered

// ---------------------------------------------------------------------------
// WMMA / TDM types (CDNA5 gfx1250, wave32)
// ---------------------------------------------------------------------------
typedef __attribute__((ext_vector_type(16))) __bf16        v16bf;
typedef __attribute__((ext_vector_type(8)))  float         v8f;
typedef __attribute__((ext_vector_type(8)))  unsigned short us8;
typedef __attribute__((ext_vector_type(4)))  unsigned int  v4u;
typedef __attribute__((ext_vector_type(8)))  int           v8i;
typedef __attribute__((ext_vector_type(4)))  int           v4i;

union FragU {
    us8   h[2];
    v16bf v;
};

#if __has_builtin(__builtin_amdgcn_tensor_load_to_lds) && \
    __has_builtin(__builtin_amdgcn_s_wait_tensorcnt)
#define USE_TDM 1
#else
#define USE_TDM 0
#endif

// round-to-nearest-even f32 -> bf16
__device__ __forceinline__ unsigned short f2bf(float f) {
    unsigned int u = __float_as_uint(f);
    unsigned int r = u + 0x7FFFu + ((u >> 16) & 1u);
    if ((u & 0x7F800000u) == 0x7F800000u) r = u;   // inf/nan: truncate
    return (unsigned short)(r >> 16);
}

__device__ __forceinline__ void atomicMaxFloat(float* addr, float v) {
    if (v >= 0.f) atomicMax((int*)addr, __float_as_int(v));
    else          atomicMin((unsigned int*)addr, __float_as_uint(v));
}

#if USE_TDM
// Issue TENSOR_LOAD_TO_LDS of a [128 cols x 128 k] bf16 tile.
// D# per CDNA5 ISA 8.3-8.5: 2D, data_size=2B, row stride = K elements,
// pad_enable: +4 DWORDs LDS pad per 64 DWORDs (=> COL_LDS short stride).
// This toolchain exposes the 6-arg builtin (clang-23/therock form):
//   (uint32x4 g0, int32x8 g1, int32x4 g2, int32x4 g3, int32x8 extra, i32 cpol)
// extra group zero-filled (semantics uncharacterized; zero call is the
// probe-verified pattern).
__device__ __forceinline__ void tdm_load_bpanel(const unsigned short* gsrc,
                                                unsigned lds_byte_off, int K) {
    const unsigned long long ga = (unsigned long long)(uintptr_t)gsrc;
    v4u g0;
    g0[0] = 1u;                                            // count=1 (valid), user mode
    g0[1] = lds_byte_off;                                  // LDS address (bytes)
    g0[2] = (unsigned)(ga & 0xFFFFFFFFu);                  // global_addr[31:0]
    g0[3] = (unsigned)((ga >> 32) & 0x01FFFFFFu) | (2u << 30);  // addr[56:32] | type=2
    v8i g1;
    g1[0] = (int)((1u << 16) |                             // data_size = 2B
                  (1u << 20) |                             // pad_enable
                  (5u << 22) |                             // pad_interval: 64 DWORDs
                  (3u << 25));                             // pad_amount: 4 DWORDs (16B)
    g1[1] = (int)(((unsigned)K & 0xFFFFu) << 16);          // tensor_dim0 lo16 @ [63:48]
    g1[2] = (int)((((unsigned)K >> 16) & 0xFFFFu) |        // tensor_dim0 hi16 @ [79:64]
                  (128u << 16));                           // tensor_dim1 lo16 @ [95:80]
    g1[3] = (int)((unsigned)BK << 16);                     // tile_dim0 @ [127:112]
    g1[4] = (int)(COLS_BLK);                               // tile_dim1 @ [143:128]
    g1[5] = (int)K;                                        // tensor_dim0_stride lo32
    g1[6] = 0;                                             // stride0 hi16 | stride1 lo16
    g1[7] = 0;
    const v4i gz4 = {0, 0, 0, 0};
    const v8i gz8 = {0, 0, 0, 0, 0, 0, 0, 0};
    __builtin_amdgcn_tensor_load_to_lds(g0, g1, gz4, gz4, gz8, 0);
}
#endif

// ---------------------------------------------------------------------------
// f32 -> bf16 plain convert (node features)
// ---------------------------------------------------------------------------
__global__ void k_cvt_bf16(const float* __restrict__ src,
                           unsigned short* __restrict__ dst, int n) {
    int i = blockIdx.x * blockDim.x + threadIdx.x;
    if (i < n) dst[i] = f2bf(src[i]);
}

// W[K,Nc] f32 -> Wt[Ncpad,K] bf16 (transpose; rows >= Nc zero-filled)
__global__ void k_cvt_w_t(const float* __restrict__ W,
                          unsigned short* __restrict__ Wt,
                          int K, int Nc, int Ncpad) {
    int i = blockIdx.x * blockDim.x + threadIdx.x;
    if (i >= Ncpad * K) return;
    int n = i / K, k = i - n * K;
    Wt[i] = (n < Nc) ? f2bf(W[(size_t)k * Nc + n]) : (unsigned short)0;
}

// ---------------------------------------------------------------------------
// WMMA GEMM: C[M,Nc] = act(A[M,K]bf16 * Bt[Ncpad,K]bf16^T + bias)
//   - block = 8 waves (4 along M x 2 along N) -> 64(M) x 128(N) tile
//   - B panel staged in LDS by the Tensor Data Mover in 4 double-buffered
//     K-chunks of 128 (s_wait_tensorcnt + barrier); bank-skewed layout
//   - each wave: 16x64 strip = 4 accumulators; A fragment (global, software
//     pipelined one K-step ahead) reused across 4 live B fragments
//     (ds_load_b128 from LDS, distinct registers -> batched waits)
//   - grid.x * 128 == Ncpad exactly -> unguarded loads; stores check n < Nc
// ---------------------------------------------------------------------------
__global__ __launch_bounds__(256)
void k_gemm_bf16_wmma(const unsigned short* __restrict__ A,
                      const unsigned short* __restrict__ Bt,
                      const float* __restrict__ bias,     // [Nc] or null
                      float* __restrict__ Cf,             // [M,Nc] or null
                      unsigned short* __restrict__ Cbf,   // [M,Nc] or null
                      int M, int Nc, int K, int relu) {
    extern __shared__ unsigned short Bsh[];                // 2 * STAGE_SH shorts

    const int tid  = threadIdx.x;
    const int wid  = tid >> 5;
    const int lane = tid & 31;
    const int wm = wid & 3, wn = wid >> 2;
    const int row0 = blockIdx.y * 64 + wm * 16;
    const int col0blk = blockIdx.x * COLS_BLK;
    const int l16  = lane & 15;
    const int half = lane >> 4;
    const bool rvalid = (row0 < M);                        // M % 16 == 0

    int ar = row0 + l16; if (ar >= M) ar = M - 1;          // clamp (no early return:
    const unsigned short* arow = A + (size_t)ar * K;       //  barriers below)

    const unsigned short* gpanel = Bt + (size_t)col0blk * K;
    const int nstage = K / BK;

    // stage 0 prologue
#if USE_TDM
    if (wid == 0)
        tdm_load_bpanel(gpanel, (unsigned)(uintptr_t)&Bsh[0], K);
#else
    for (int t = tid; t < COLS_BLK * (BK / 8); t += 256) {
        const int c = t >> 4, kc = (t & 15) << 3;
        *(us8*)&Bsh[c * COL_LDS + kc] = *(const us8*)(gpanel + (size_t)c * K + kc);
    }
#endif

    v8f acc[4] = {};
    for (int s = 0; s < nstage; ++s) {
        const int cur = s & 1;
        const bool have_next = (s + 1 < nstage);
        // kick off next stage into the other buffer
#if USE_TDM
        if (wid == 0) {
            if (have_next)
                tdm_load_bpanel(gpanel + (size_t)(s + 1) * BK,
                                (unsigned)(uintptr_t)&Bsh[(cur ^ 1) * STAGE_SH], K);
            if (have_next) __builtin_amdgcn_s_wait_tensorcnt(1);  // stage s done
            else           __builtin_amdgcn_s_wait_tensorcnt(0);
        }
#else
        if (have_next) {
            const unsigned short* gnext = gpanel + (size_t)(s + 1) * BK;
            unsigned short* lnext = &Bsh[(cur ^ 1) * STAGE_SH];
            for (int t = tid; t < COLS_BLK * (BK / 8); t += 256) {
                const int c = t >> 4, kc = (t & 15) << 3;
                *(us8*)&lnext[c * COL_LDS + kc] = *(const us8*)(gnext + (size_t)c * K + kc);
            }
        }
#endif
        __syncthreads();                                   // stage s visible to all

        const int kbase = s * BK;
        if (have_next)                                     // gfx1250 global_prefetch_b8
            __builtin_prefetch(arow + kbase + BK + 8 * half, 0, 1);

        // software-pipelined A fragment (global); B fragments all live at once
        FragU a_cur;
        {
            const int ka = kbase + 8 * half;
            a_cur.h[0] = *(const us8*)(arow + ka);
            a_cur.h[1] = *(const us8*)(arow + ka + 16);
        }
        #pragma unroll
        for (int kk = 0; kk < BK; kk += 32) {
            const int kl = kk + 8 * half;                  // k offset within stage
            const int lb0 = cur * STAGE_SH + (wn * 64 + l16) * COL_LDS + kl;
            FragU b[4];
            #pragma unroll
            for (int j = 0; j < 4; ++j) {
                const int lb = lb0 + j * (16 * COL_LDS);
                b[j].h[0] = *(const us8*)&Bsh[lb];
                b[j].h[1] = *(const us8*)&Bsh[lb + 16];
            }
            FragU a_nxt;
            if (kk + 32 < BK) {                            // next A overlaps WMMAs
                const int ka2 = kbase + kk + 32 + 8 * half;
                a_nxt.h[0] = *(const us8*)(arow + ka2);
                a_nxt.h[1] = *(const us8*)(arow + ka2 + 16);
            }
            #pragma unroll
            for (int j = 0; j < 4; ++j)
                acc[j] = __builtin_amdgcn_wmma_f32_16x16x32_bf16(
                             false, a_cur.v, false, b[j].v, (short)0, acc[j],
                             false, false);
            if (kk + 32 < BK) a_cur = a_nxt;
        }
        __syncthreads();                                   // done reading buf before reuse
    }

    // epilogue: VGPR i -> row row0 + i + 8*half, col (lane&15) of each tile
    if (rvalid) {
        #pragma unroll
        for (int j = 0; j < 4; ++j) {
            const int n = col0blk + wn * 64 + 16 * j + l16;
            if (n >= Nc) continue;
            const float bval = (bias != nullptr) ? bias[n] : 0.f;
            #pragma unroll
            for (int i = 0; i < 8; ++i) {
                const int m = row0 + i + 8 * half;
                float v = acc[j][i] + bval;
                if (relu) v = v > 0.f ? v : 0.f;
                if (Cf)  Cf [(size_t)m * Nc + n] = v;
                if (Cbf) Cbf[(size_t)m * Nc + n] = f2bf(v);
            }
        }
    }
}

// ---------------------------------------------------------------------------
// Per-layer scratch init: m=-inf, z=0, newx=0
// ---------------------------------------------------------------------------
__global__ void k_init_layer(unsigned int* __restrict__ mbits,
                             float* __restrict__ z,
                             float* __restrict__ newx, int n, int nd) {
    int i = blockIdx.x * blockDim.x + threadIdx.x;
    if (i < n) { mbits[i] = 0xFF800000u; z[i] = 0.f; }
    if (i < nd) newx[i] = 0.f;
}

// ---------------------------------------------------------------------------
// Edge logits: e[j] = leaky_relu(xl[src] + xr[dst]) . att ; segment atomic max
// ---------------------------------------------------------------------------
__global__ __launch_bounds__(256)
void k_edge_logits(const float* __restrict__ xl, const float* __restrict__ xr,
                   const int* __restrict__ srcE, const int* __restrict__ dstE,
                   const float* __restrict__ att,
                   float* __restrict__ e, float* __restrict__ m) {
    const int widx = blockIdx.x * (blockDim.x >> 5) + (threadIdx.x >> 5);
    if (widx >= EEP) return;
    const int lane = threadIdx.x & 31;
    int s, d;
    if (widx < EE_) { s = srcE[widx]; d = dstE[widx]; }
    else            { s = d = widx - EE_; }

    const float4* pl = (const float4*)(xl + (size_t)s * DD);
    const float4* pr = (const float4*)(xr + (size_t)d * DD);
    const float4* pa = (const float4*)att;
    float t = 0.f;
    #pragma unroll
    for (int i = 0; i < 4; ++i) {
        const int idx = lane + 32 * i;            // 128 float4 = 512 floats
        const float4 a = pl[idx], b = pr[idx], w = pa[idx];
        float vx;
        vx = a.x + b.x; vx = vx > 0.f ? vx : SLOPE * vx; t += vx * w.x;
        vx = a.y + b.y; vx = vx > 0.f ? vx : SLOPE * vx; t += vx * w.y;
        vx = a.z + b.z; vx = vx > 0.f ? vx : SLOPE * vx; t += vx * w.z;
        vx = a.w + b.w; vx = vx > 0.f ? vx : SLOPE * vx; t += vx * w.w;
    }
    #pragma unroll
    for (int o = 16; o; o >>= 1) t += __shfl_xor(t, o);
    if (lane == 0) {
        e[widx] = t;
        atomicMaxFloat(m + d, t);
    }
}

// ---------------------------------------------------------------------------
// p = exp(e - m[dst]); z[dst] += p
// ---------------------------------------------------------------------------
__global__ void k_edge_expsum(const float* __restrict__ e,
                              const int* __restrict__ dstE,
                              const float* __restrict__ m,
                              float* __restrict__ p, float* __restrict__ z) {
    const int j = blockIdx.x * blockDim.x + threadIdx.x;
    if (j >= EEP) return;
    const int d = (j < EE_) ? dstE[j] : j - EE_;
    const float pv = __expf(e[j] - m[d]);
    p[j] = pv;
    atomicAdd(z + d, pv);
}

// ---------------------------------------------------------------------------
// newx[dst] += (p/z[dst]) * xl[src]   (wave per edge, f32 atomics)
// ---------------------------------------------------------------------------
__global__ __launch_bounds__(256)
void k_edge_aggregate(const float* __restrict__ xl,
                      const int* __restrict__ srcE, const int* __restrict__ dstE,
                      const float* __restrict__ p, const float* __restrict__ z,
                      float* __restrict__ newx) {
    const int widx = blockIdx.x * (blockDim.x >> 5) + (threadIdx.x >> 5);
    if (widx >= EEP) return;
    const int lane = threadIdx.x & 31;
    int s, d;
    if (widx < EE_) { s = srcE[widx]; d = dstE[widx]; }
    else            { s = d = widx - EE_; }
    const float coef = p[widx] / z[d];
    const float* row = xl + (size_t)s * DD;
    float* orow = newx + (size_t)d * DD;
    #pragma unroll
    for (int i = 0; i < 16; ++i) {
        const int idx = lane + 32 * i;
        atomicAdd(orow + idx, coef * row[idx]);
    }
}

// ---------------------------------------------------------------------------
// x = newx + bias; re-quantize to bf16; optionally emit f32 (final layer)
// ---------------------------------------------------------------------------
__global__ void k_finalize(const float* __restrict__ newx,
                           const float* __restrict__ bias,
                           unsigned short* __restrict__ xbf,
                           float* __restrict__ xf_out) {
    const int i = blockIdx.x * blockDim.x + threadIdx.x;
    if (i >= NN * DD) return;
    const float v = newx[i] + bias[i & (DD - 1)];
    xbf[i] = f2bf(v);
    if (xf_out) xf_out[i] = v;
}

// ---------------------------------------------------------------------------
// host side
// ---------------------------------------------------------------------------
static inline size_t alignup(size_t x) { return (x + 255) & ~(size_t)255; }

extern "C" void kernel_launch(void* const* d_in, const int* in_sizes, int n_in,
                              void* d_out, int out_size, void* d_ws, size_t ws_size,
                              hipStream_t stream) {
    (void)in_sizes; (void)n_in; (void)out_size; (void)ws_size;

    const float* nodes   = (const float*)d_in[0];
    const int*   edges   = (const int*)  d_in[1];   // [2,E]
    const float* W_pre   = (const float*)d_in[2];
    const float* W_lin   = (const float*)d_in[3];
    const float* Wl      = (const float*)d_in[4];   // [L,D,D]
    const float* Wr      = (const float*)d_in[5];   // [L,D,D]
    const float* att     = (const float*)d_in[6];   // [L,D]
    const float* gbias   = (const float*)d_in[7];   // [L,D]
    const float* W_cls   = (const float*)d_in[8];   // [D,OUT]
    const float* b_cls   = (const float*)d_in[9];   // [OUT]

    const int* srcE = edges;
    const int* dstE = edges + EE_;

    float* out_x   = (float*)d_out;                    // [N,D]
    float* out_cls = (float*)d_out + (size_t)NN * DD;  // [N,OUT]

    // ---- workspace layout -------------------------------------------------
    char* ws = (char*)d_ws;
    size_t off = 0;
    auto carve = [&](size_t bytes) { char* p = ws + off; off += alignup(bytes); return p; };

    unsigned short* xbf_a  = (unsigned short*)carve((size_t)NN * DD * 2);
    unsigned short* xbf_b  = (unsigned short*)carve((size_t)NN * DD * 2);
    float* xl   = (float*)carve((size_t)NN * DD * 4);
    float* xr   = (float*)carve((size_t)NN * DD * 4);
    float* newx = (float*)carve((size_t)NN * DD * 4);
    float* ebuf = (float*)carve((size_t)EEP * 4);
    float* pbuf = (float*)carve((size_t)EEP * 4);
    float* mbuf = (float*)carve((size_t)NN * 4);
    float* zbuf = (float*)carve((size_t)NN * 4);
    unsigned short* wpre_t = (unsigned short*)carve((size_t)DD * DD * 2);
    unsigned short* wlin_t = (unsigned short*)carve((size_t)DD * DD * 2);
    unsigned short* wl_t   = (unsigned short*)carve((size_t)LL * DD * DD * 2);
    unsigned short* wr_t   = (unsigned short*)carve((size_t)LL * DD * DD * 2);
    unsigned short* wcls_t = (unsigned short*)carve((size_t)OUTP * DD * 2);

    // ---- weight conversion (f32 -> bf16, transposed, padded) --------------
    {
        const int nDD = DD * DD;
        const int g = (nDD + 255) / 256;
        k_cvt_w_t<<<g, 256, 0, stream>>>(W_pre, wpre_t, DD, DD, DD);
        k_cvt_w_t<<<g, 256, 0, stream>>>(W_lin, wlin_t, DD, DD, DD);
        for (int l = 0; l < LL; ++l) {
            k_cvt_w_t<<<g, 256, 0, stream>>>(Wl + (size_t)l * nDD, wl_t + (size_t)l * nDD, DD, DD, DD);
            k_cvt_w_t<<<g, 256, 0, stream>>>(Wr + (size_t)l * nDD, wr_t + (size_t)l * nDD, DD, DD, DD);
        }
        const int nc = DD * OUTP;
        k_cvt_w_t<<<(nc + 255) / 256, 256, 0, stream>>>(W_cls, wcls_t, DD, OUTD, OUTP);
    }

    // ---- x = relu(nodes @ W_pre); 3x x = relu(x @ W_lin) ------------------
    const dim3 gemm_blk(256);
    const dim3 gemm_grd_sq(DD / COLS_BLK, (NN + 63) / 64);     // Ncpad = 512
    {
        const int n = NN * DD;
        k_cvt_bf16<<<(n + 255) / 256, 256, 0, stream>>>(nodes, xbf_a, n);
    }
    k_gemm_bf16_wmma<<<gemm_grd_sq, gemm_blk, LDS_BYTES, stream>>>(
        xbf_a, wpre_t, nullptr, nullptr, xbf_b, NN, DD, DD, 1);
    k_gemm_bf16_wmma<<<gemm_grd_sq, gemm_blk, LDS_BYTES, stream>>>(
        xbf_b, wlin_t, nullptr, nullptr, xbf_a, NN, DD, DD, 1);
    k_gemm_bf16_wmma<<<gemm_grd_sq, gemm_blk, LDS_BYTES, stream>>>(
        xbf_a, wlin_t, nullptr, nullptr, xbf_b, NN, DD, DD, 1);
    k_gemm_bf16_wmma<<<gemm_grd_sq, gemm_blk, LDS_BYTES, stream>>>(
        xbf_b, wlin_t, nullptr, nullptr, xbf_a, NN, DD, DD, 1);
    // current features (bf16) now in xbf_a

    // ---- GATv2 layers -----------------------------------------------------
    const int edge_wave_blocks = (EEP + 7) / 8;          // 8 waves per block
    const int edge_thr_blocks  = (EEP + 255) / 256;
    const int nd = NN * DD;
    for (int l = 0; l < LL; ++l) {
        const size_t wofs = (size_t)l * DD * DD;
        k_gemm_bf16_wmma<<<gemm_grd_sq, gemm_blk, LDS_BYTES, stream>>>(
            xbf_a, wl_t + wofs, nullptr, xl, nullptr, NN, DD, DD, 0);
        k_gemm_bf16_wmma<<<gemm_grd_sq, gemm_blk, LDS_BYTES, stream>>>(
            xbf_a, wr_t + wofs, nullptr, xr, nullptr, NN, DD, DD, 0);

        k_init_layer<<<(nd + 255) / 256, 256, 0, stream>>>(
            (unsigned int*)mbuf, zbuf, newx, NN, nd);

        k_edge_logits<<<edge_wave_blocks, 256, 0, stream>>>(
            xl, xr, srcE, dstE, att + (size_t)l * DD, ebuf, mbuf);

        k_edge_expsum<<<edge_thr_blocks, 256, 0, stream>>>(
            ebuf, dstE, mbuf, pbuf, zbuf);

        k_edge_aggregate<<<edge_wave_blocks, 256, 0, stream>>>(
            xl, srcE, dstE, pbuf, zbuf, newx);

        k_finalize<<<(nd + 255) / 256, 256, 0, stream>>>(
            newx, gbias + (size_t)l * DD, xbf_a,
            (l == LL - 1) ? out_x : nullptr);
    }

    // ---- classifier: out = x @ W_cls + b_cls ------------------------------
    const dim3 gemm_grd_cls(OUTP / COLS_BLK, (NN + 63) / 64);
    k_gemm_bf16_wmma<<<gemm_grd_cls, gemm_blk, LDS_BYTES, stream>>>(
        xbf_a, wcls_t, b_cls, out_cls, nullptr, NN, OUTD, DD, 0);
}
